// TransMatcher_14070312861937
// MI455X (gfx1250) — matrix operand order repaired and checked by reading, not compiled, not run on hardware
//
#include <hip/hip_runtime.h>
#include <hip/hip_bf16.h>

// ---------------- problem constants ----------------
#define NQ   48
#define NK   48
#define NL   3
#define DD   512
#define SS   192
#define DFF  2048
#define QPROWS (NQ * SS)          // 9216 rows per side per layer
#define MTOT   (2 * QPROWS)       // 18432 combined rows (memory then features)
#define BNS  0.9999950000374996f  // 1/sqrt(1+1e-5)

// ---------------- WMMA plumbing ----------------
typedef __attribute__((ext_vector_type(16))) __bf16    bf16x16;
typedef __attribute__((ext_vector_type(8)))  float     f32x8;
typedef __attribute__((ext_vector_type(4)))  unsigned  u32x4;

union Frag {
    bf16x16 bf;
    u32x4   q[2];
};

__device__ __forceinline__ f32x8 wmma_bf16(const Frag& a, const Frag& b, f32x8 c) {
    // v_wmma_f32_16x16x32_bf16: D = A(16x32) * B(32x16) + C, fp32 accumulate
    return __builtin_amdgcn_wmma_f32_16x16x32_bf16(
        /*neg_a=*/false, a.bf, /*neg_b=*/false, b.bf,
        /*c_mod=*/(short)0, c, /*reuse_a=*/false, /*reuse_b=*/false);
}

// native hardware fp32->bf16 conversion (let clang pick packed cvt forms)
__device__ __forceinline__ void pack8(Frag& f, int base, float4 a, float4 b) {
    f.bf[base + 0] = (__bf16)a.x; f.bf[base + 1] = (__bf16)a.y;
    f.bf[base + 2] = (__bf16)a.z; f.bf[base + 3] = (__bf16)a.w;
    f.bf[base + 4] = (__bf16)b.x; f.bf[base + 5] = (__bf16)b.y;
    f.bf[base + 6] = (__bf16)b.z; f.bf[base + 7] = (__bf16)b.w;
}

// order-preserving float<->uint map for atomicMax on floats
__device__ __forceinline__ unsigned f2ord(float f) {
    unsigned u = __float_as_uint(f);
    return (u & 0x80000000u) ? ~u : (u | 0x80000000u);
}
__device__ __forceinline__ float ord2f(unsigned u) {
    return __uint_as_float((u & 0x80000000u) ? (u & 0x7FFFFFFFu) : ~u);
}

// ---------------- kernel 1: sigmoid(gate) + zero output ----------------
__global__ __launch_bounds__(256) void prep_kernel(const float* __restrict__ se,
                                                   float* __restrict__ gate,
                                                   float* __restrict__ out) {
    int i = blockIdx.x * 256 + threadIdx.x;
    if (i < NL * SS * SS) gate[i] = 1.0f / (1.0f + __expf(-se[i]));
    if (i < NQ * NK)      out[i] = 0.0f;
}

// ---------------- kernel 2: projection GEMM (fp32 in -> bf16 out) ----------------
// Combined M = [memory rows | features rows] per layer; N = 512 out-features.
// grid = (8 n-tiles of 64, 144 m-tiles of 128, 3 layers); 8 waves, wave = 32x32.
__global__ __launch_bounds__(256) void proj_kernel(const float* __restrict__ mem,
                                                   const float* __restrict__ feat,
                                                   const float* __restrict__ w1,
                                                   const float* __restrict__ b1,
                                                   __bf16* __restrict__ Qp,
                                                   __bf16* __restrict__ Kp) {
    const int lane = threadIdx.x & 31;
    const int wave = threadIdx.x >> 5;
    const int l15 = lane & 15, half = lane >> 4;
    const int layer = blockIdx.z;
    const int m0 = blockIdx.y * 128 + (wave >> 1) * 32;   // combined-row base
    const int n0 = blockIdx.x * 64  + (wave & 1)  * 32;   // out-feature base
    const int side = m0 / QPROWS;                         // 0: memory->Qp, 1: features->Kp
    const int r0 = m0 - side * QPROWS;                    // side-relative row
    const float* src = side ? feat : mem;
    const float* wB  = w1 + (size_t)layer * DD * DD;

    f32x8 acc[2][2];
#pragma unroll
    for (int i = 0; i < 2; ++i)
#pragma unroll
        for (int j = 0; j < 2; ++j) acc[i][j] = (f32x8)0.0f;

    for (int k0 = 0; k0 < DD; k0 += 32) {
        Frag fa[2], fb[2];
#pragma unroll
        for (int i = 0; i < 2; ++i) {
            // A fragment: rows of X (stride 1536 fp32, layer chunk offset)
            // lane<16 holds K{0..7},{16..23}; lane>=16 holds K{8..15},{24..31}
            const float* ap = src + (size_t)(r0 + i * 16 + l15) * (DD * NL)
                                  + (size_t)layer * DD + k0 + 8 * half;
            pack8(fa[i], 0, *(const float4*)(ap + 0),  *(const float4*)(ap + 4));
            pack8(fa[i], 8, *(const float4*)(ap + 16), *(const float4*)(ap + 20));
            // B fragment: B[k][n] = W1[n][k]; lane = n, lane<16 holds K{0..7},{16..23}; lane>=16 holds K{8..15},{24..31}
            const float* bp = wB + (size_t)(n0 + i * 16 + l15) * DD + k0 + 8 * half;
            pack8(fb[i], 0, *(const float4*)(bp + 0), *(const float4*)(bp + 4));
            pack8(fb[i], 8, *(const float4*)(bp + 16), *(const float4*)(bp + 20));
        }
#pragma unroll
        for (int i = 0; i < 2; ++i)
#pragma unroll
            for (int j = 0; j < 2; ++j) acc[i][j] = wmma_bf16(fa[i], fb[j], acc[i][j]);
    }

    __bf16* dst = (side ? Kp : Qp) + (size_t)layer * QPROWS * DD;
#pragma unroll
    for (int j = 0; j < 2; ++j) {
        const int col = n0 + j * 16 + l15;
        const float bias = b1[layer * DD + col];
#pragma unroll
        for (int i = 0; i < 2; ++i)
#pragma unroll
            for (int r = 0; r < 8; ++r) {
                const int row = r0 + i * 16 + r + 8 * half;  // C layout: M = r + 8*half
                dst[(size_t)row * DD + col] = (__bf16)(acc[i][j][r] + bias);
            }
    }
}

// ---------------- kernel 3: fused pair-score + gate + dual max + MLP head ----------------
// grid = (k=48, q=48, layer=3); 512 threads = 16 waves, each wave a 48x48 tile of the
// 192x192 score matrix. The [q,k,s,t] tensor never touches memory.
__global__ __launch_bounds__(512) void score_kernel(
    const __bf16* __restrict__ Qp, const __bf16* __restrict__ Kp,
    const float* __restrict__ gate,
    const float* __restrict__ w2,  const float* __restrict__ b2,
    const float* __restrict__ g2,  const float* __restrict__ bb2,
    const float* __restrict__ w3,  const float* __restrict__ b3,
    const float* __restrict__ bn1g, const float* __restrict__ bn1b,
    const float* __restrict__ bn3g, const float* __restrict__ bn3b,
    float* __restrict__ out) {

    const int tid = threadIdx.x, lane = tid & 31, wave = tid >> 5;
    const int l15 = lane & 15, half = lane >> 4;
    const int layer = blockIdx.z, kk = blockIdx.x, qq = blockIdx.y;
    const int s0 = (wave >> 2) * 48, t0 = (wave & 3) * 48;

    __shared__ unsigned smax[2][SS];   // [0]: max over s (per t) ; [1]: max over t (per s)
    __shared__ float    xr[2][SS];
    __shared__ float    wsum[16];
    if (tid < 2 * SS) smax[tid / SS][tid % SS] = 0u;   // ordered-key "-inf"
    __syncthreads();

    const __bf16* Abase = Kp + ((size_t)layer * QPROWS + (size_t)kk * SS) * DD;
    const __bf16* Bbase = Qp + ((size_t)layer * QPROWS + (size_t)qq * SS) * DD;

    f32x8 acc[3][3];
#pragma unroll
    for (int i = 0; i < 3; ++i)
#pragma unroll
        for (int j = 0; j < 3; ++j) acc[i][j] = (f32x8)0.0f;

    for (int k0 = 0; k0 < DD; k0 += 32) {
        Frag fa[3], fb[3];
#pragma unroll
        for (int i = 0; i < 3; ++i) {
            const __bf16* ap = Abase + (size_t)(s0 + i * 16 + l15) * DD + k0 + 8 * half;
            fa[i].q[0] = *(const u32x4*)(ap);
            fa[i].q[1] = *(const u32x4*)(ap + 16);
            const __bf16* bp = Bbase + (size_t)(t0 + i * 16 + l15) * DD + k0 + 8 * half;
            fb[i].q[0] = *(const u32x4*)(bp);
            fb[i].q[1] = *(const u32x4*)(bp + 16);
        }
#pragma unroll
        for (int i = 0; i < 3; ++i)
#pragma unroll
            for (int j = 0; j < 3; ++j) acc[i][j] = wmma_bf16(fa[i], fb[j], acc[i][j]);
    }

    // epilogue: gate multiply + dual max (in regs -> shuffles -> LDS atomicMax)
    const float* g = gate + (size_t)layer * SS * SS;
    float cmax[3] = {-INFINITY, -INFINITY, -INFINITY};
#pragma unroll
    for (int i = 0; i < 3; ++i) {
#pragma unroll
        for (int r = 0; r < 8; ++r) {
            const int s = s0 + i * 16 + r + 8 * half;
            float rmax = -INFINITY;
#pragma unroll
            for (int j = 0; j < 3; ++j) {
                const int t = t0 + j * 16 + l15;
                const float v = acc[i][j][r] * g[s * SS + t];
                rmax = fmaxf(rmax, v);
                cmax[j] = fmaxf(cmax[j], v);
            }
#pragma unroll
            for (int m = 1; m < 16; m <<= 1) rmax = fmaxf(rmax, __shfl_xor(rmax, m, 32));
            if (l15 == 0) atomicMax(&smax[1][s], f2ord(rmax));
        }
    }
#pragma unroll
    for (int j = 0; j < 3; ++j) {
        float cm = fmaxf(cmax[j], __shfl_xor(cmax[j], 16, 32));
        if (half == 0) atomicMax(&smax[0][t0 + j * 16 + l15], f2ord(cm));
    }
    __syncthreads();

    // BN1 (scalar affine per layer) into xr
    const float g1s = bn1g[layer] * BNS, b1s = bn1b[layer];
    if (tid < 2 * SS) {
        const int j = tid / SS, idx = tid % SS;
        xr[j][idx] = ord2f(smax[j][idx]) * g1s + b1s;
    }
    __syncthreads();

    // fc2 -> BN2 -> ReLU -> fc3 (fused, no hidden materialization)
    float part = 0.0f;
    const float* w2l = w2 + (size_t)layer * DFF * SS;
    const float* b2l = b2 + (size_t)layer * DFF;
    const float* g2l = g2 + (size_t)layer * DFF;
    const float* bb2l = bb2 + (size_t)layer * DFF;
    const float* w3l = w3 + (size_t)layer * DFF;
    for (int it = 0; it < 8; ++it) {
        const int gi = tid + it * 512;        // 0..4095 over (j, f)
        const int j = gi >> 11, f = gi & (DFF - 1);
        const float* wr = w2l + (size_t)f * SS;
        float h = b2l[f];
#pragma unroll 8
        for (int t = 0; t < SS; ++t) h = fmaf(xr[j][t], wr[t], h);
        h = h * (g2l[f] * BNS) + bb2l[f];
        h = fmaxf(h, 0.0f);
        part += h * w3l[f];
    }
#pragma unroll
    for (int m = 1; m < 32; m <<= 1) part += __shfl_xor(part, m, 32);
    if (lane == 0) wsum[wave] = part;
    __syncthreads();
    if (tid == 0) {
        float tot = 0.0f;
#pragma unroll
        for (int w = 0; w < 16; ++w) tot += wsum[w];
        tot += 2.0f * b3[layer];                              // y0+y1, each + b3
        const float contrib = tot * (bn3g[layer] * BNS) + bn3b[layer];
        atomicAdd(&out[qq * NK + kk], contrib);
    }
}

// ---------------- kernel 4: final BN over [Q,K] ----------------
__global__ __launch_bounds__(256) void final_kernel(float* __restrict__ out,
                                                    const float* __restrict__ ng,
                                                    const float* __restrict__ nb) {
    int i = blockIdx.x * 256 + threadIdx.x;
    if (i < NQ * NK) out[i] = out[i] * (ng[0] * BNS) + nb[0];
}

// ---------------- launch ----------------
extern "C" void kernel_launch(void* const* d_in, const int* in_sizes, int n_in,
                              void* d_out, int out_size, void* d_ws, size_t ws_size,
                              hipStream_t stream) {
    const float* memory   = (const float*)d_in[0];
    const float* features = (const float*)d_in[1];
    const float* fc1_w    = (const float*)d_in[2];
    const float* fc1_b    = (const float*)d_in[3];
    const float* se       = (const float*)d_in[4];
    const float* bn1_g    = (const float*)d_in[5];
    const float* bn1_b    = (const float*)d_in[6];
    const float* fc2_w    = (const float*)d_in[7];
    const float* fc2_b    = (const float*)d_in[8];
    const float* bn2_g    = (const float*)d_in[9];
    const float* bn2_b    = (const float*)d_in[10];
    const float* fc3_w    = (const float*)d_in[11];
    const float* fc3_b    = (const float*)d_in[12];
    const float* bn3_g    = (const float*)d_in[13];
    const float* bn3_b    = (const float*)d_in[14];
    const float* norm_g   = (const float*)d_in[15];
    const float* norm_b   = (const float*)d_in[16];
    float* out = (float*)d_out;

    char* ws = (char*)d_ws;
    float* gate = (float*)ws;
    size_t off = (size_t)NL * SS * SS * sizeof(float);             // 442 KB
    __bf16* Qp = (__bf16*)(ws + off);
    off += (size_t)NL * QPROWS * DD * sizeof(__bf16);              // 28.3 MB
    __bf16* Kp = (__bf16*)(ws + off);                              // 28.3 MB

    prep_kernel<<<dim3((NL * SS * SS + 255) / 256), 256, 0, stream>>>(se, gate, out);
    proj_kernel<<<dim3(DD / 64, MTOT / 128, NL), 256, 0, stream>>>(
        memory, features, fc1_w, fc1_b, Qp, Kp);
    score_kernel<<<dim3(NK, NQ, NL), 512, 0, stream>>>(
        Qp, Kp, gate, fc2_w, fc2_b, bn2_g, bn2_b, fc3_w, fc3_b,
        bn1_g, bn1_b, bn3_g, bn3_b, out);
    final_kernel<<<dim3((NQ * NK + 255) / 256), 256, 0, stream>>>(out, norm_g, norm_b);
}
